// AdaptiveGatedCrossAttention_6734508720599
// MI455X (gfx1250) — matrix operand order repaired and checked by reading, not compiled, not run on hardware
//
#include <hip/hip_runtime.h>
#include <hip/hip_bf16.h>
#include <math.h>

#define DEV __device__ __forceinline__

typedef __bf16 bf16;
typedef __bf16 v16bf __attribute__((ext_vector_type(16)));
typedef float  v8f   __attribute__((ext_vector_type(8)));
typedef unsigned int v4u __attribute__((ext_vector_type(4)));
typedef int v4i __attribute__((ext_vector_type(4)));
typedef int v8i __attribute__((ext_vector_type(8)));

static constexpr int B_  = 16;
static constexpr int S_  = 4096;
static constexpr int H_  = 768;
static constexpr int C_  = 64;
static constexpr int NH_ = 8;
static constexpr int HD_ = 96;
static constexpr int M_  = B_ * S_;    // 65536
static constexpr int KG_ = 2 * H_;     // 1536 (gate GEMM K after rank-1 trick)

#if __has_builtin(__builtin_amdgcn_tensor_load_to_lds) && __has_builtin(__builtin_amdgcn_s_wait_tensorcnt)
#define USE_TDM 1
#else
#define USE_TDM 0
#endif

union Frag { v16bf v; uint4 q[2]; };

DEV v8f vzero() { v8f z = {0.f,0.f,0.f,0.f,0.f,0.f,0.f,0.f}; return z; }

DEV v8f wmma_bf16(v16bf a, v16bf b, v8f c) {
  // D = A(16x32 bf16) * B(32x16 bf16) + C(f32)
  return __builtin_amdgcn_wmma_f32_16x16x32_bf16(false, a, false, b, (short)0, c, false, false);
}

// A fragment: p = &A[row(lane&15)][kstep]; k0 = (lane>>4)*8
DEV v16bf load_a(const bf16* p, int k0) {
  Frag f;
  f.q[0] = *(const uint4*)(p + 2 * k0);
  f.q[1] = *(const uint4*)(p + 2 * k0 + 8);
  return f.v;
}
// B fragment: p = &WT[col][kstep + (lane>>4)*16]; 16 contiguous bf16
DEV v16bf load_b(const bf16* p) {
  Frag f;
  f.q[0] = *(const uint4*)(p);
  f.q[1] = *(const uint4*)(p + 8);
  return f.v;
}
// A fragment built on the fly from f32 source
DEV v16bf cvt_a_f32(const float* p, int k0) {
  v16bf r;
  const float* q0 = p + 2 * k0;
  const float* q1 = p + 2 * k0 + 8;
#pragma unroll
  for (int i = 0; i < 8; ++i) { r[i] = (bf16)q0[i]; r[8 + i] = (bf16)q1[i]; }
  return r;
}

// ---------------- TDM-staged B panel (gate GEMM) ----------------
// LDS image of one 16-col x 1536-K bf16 tile, with a 16B pad every 1024 data
// bytes (TDM pad_interval=256 DW, pad_amount=4 DW) to rotate banks.
static constexpr int TILE_DATA = 16 * KG_ * 2;                 // 49152 B
static constexpr int TILE_PAD  = TILE_DATA + (TILE_DATA / 1024) * 16;  // 49920 B

DEV int pad_off(int b) { return b + ((b >> 10) << 4); }

// padded-LDS B fragment: b = column_row_base + k-byte offset (16B aligned)
DEV v16bf load_b_lds(const char* base, int b) {
  Frag f;
  f.q[0] = *(const uint4*)(base + pad_off(b));
  f.q[1] = *(const uint4*)(base + pad_off(b + 16));
  return f.v;
}

#if USE_TDM
DEV void tdm_load_tile(unsigned int lds_addr, const bf16* gptr) {
  union { v4u v; unsigned int w[4]; } g0;
  union { v8i v; unsigned int w[8]; } g1;
  const unsigned long long ga = (unsigned long long)(uintptr_t)gptr;
  g0.w[0] = 1u;                                   // count=1, user descriptor
  g0.w[1] = lds_addr;                             // LDS byte address
  g0.w[2] = (unsigned int)ga;                     // global_addr[31:0]
  g0.w[3] = (unsigned int)(ga >> 32) | 0x80000000u;  // addr[56:32] | type=2
  // data_size=2B | pad_enable | pad_interval=256DW | pad_amount=4DW
  g1.w[0] = (1u << 16) | (1u << 20) | (7u << 22) | (3u << 25);
  g1.w[1] = (unsigned int)(KG_ & 0xFFFF) << 16;   // tensor_dim0=1536 (lo16)
  g1.w[2] = (unsigned int)(768 & 0xFFFF) << 16;   // dim0 hi=0 | tensor_dim1=768 (lo16)
  g1.w[3] = (unsigned int)(KG_ & 0xFFFF) << 16;   // dim1 hi=0 | tile_dim0=1536
  g1.w[4] = 16u;                                  // tile_dim1=16, tile_dim2=0
  g1.w[5] = (unsigned int)KG_;                    // tensor_dim0_stride=1536
  g1.w[6] = 0u;
  g1.w[7] = 0u;
  const v4i z4 = {0, 0, 0, 0};
#if defined(__clang_major__) && (__clang_major__ >= 23)
  const v8i z8 = {0, 0, 0, 0, 0, 0, 0, 0};
  __builtin_amdgcn_tensor_load_to_lds(g0.v, g1.v, z4, z4, z8, 0);
#else
  __builtin_amdgcn_tensor_load_to_lds(g0.v, g1.v, z4, z4, 0);
#endif
}
#else
DEV void coop_load_tile(char* dst, const bf16* g) {
  for (int i = threadIdx.x; i < TILE_DATA / 16; i += 256) {
    const int b = i * 16;
    *(uint4*)(dst + pad_off(b)) = *(const uint4*)((const char*)g + b);
  }
}
#endif

// ---------------- core 32x64 GEMM per wave ----------------
DEV void gemm32x64(const bf16* __restrict__ A, int lda,
                   const bf16* __restrict__ Bt, int ldb,
                   int m0, int n0, int K, int lane, v8f acc[2][4]) {
  const int rA = lane & 15;
  const int k0 = (lane >> 4) * 8;
  const int kb = (lane >> 4) * 16;
  const bf16* a0 = A + (size_t)(m0 + rA) * lda;
  const bf16* a1 = a0 + (size_t)16 * lda;
  const bf16* bc[4];
#pragma unroll
  for (int ni = 0; ni < 4; ++ni)
    bc[ni] = Bt + (size_t)(n0 + ni * 16 + rA) * ldb + kb;
  for (int k = 0; k < K; k += 32) {
    __builtin_prefetch(a0 + k + 128, 0, 1);   // global_prefetch_b8
    v16bf af0 = load_a(a0 + k, k0);
    v16bf af1 = load_a(a1 + k, k0);
#pragma unroll
    for (int ni = 0; ni < 4; ++ni) {
      v16bf bb = load_b(bc[ni] + k);
      acc[0][ni] = wmma_bf16(af0, bb, acc[0][ni]);
      acc[1][ni] = wmma_bf16(af1, bb, acc[1][ni]);
    }
  }
}

// ---------------- GEMM 1: Q projection -> bf16 [B][NH][S][HD] ----------------
__global__ void __launch_bounds__(256) k_qproj(const bf16* __restrict__ Xh,
                                               const bf16* __restrict__ WqT,
                                               const float* __restrict__ bq,
                                               bf16* __restrict__ Qh) {
  const int lane = threadIdx.x & 31;
  const int w = blockIdx.x * 8 + (threadIdx.x >> 5);   // 24576 waves
  const int m0 = (w / 12) * 32;
  const int n0 = (w % 12) * 64;
  v8f acc[2][4];
#pragma unroll
  for (int i = 0; i < 2; ++i)
#pragma unroll
    for (int j = 0; j < 4; ++j) acc[i][j] = vzero();
  gemm32x64(Xh, H_, WqT, H_, m0, n0, H_, lane, acc);
  const int hi = lane >> 4, rA = lane & 15;
#pragma unroll
  for (int mi = 0; mi < 2; ++mi)
#pragma unroll
    for (int ni = 0; ni < 4; ++ni) {
      const int n = n0 + ni * 16 + rA;
      const float bias = bq[n];
      const int h = n / HD_, d = n % HD_;
#pragma unroll
      for (int r = 0; r < 8; ++r) {
        const int m = m0 + mi * 16 + r + hi * 8;
        const int b = m >> 12, s = m & 4095;
        Qh[((((size_t)b * NH_ + h) * S_) + s) * HD_ + d] = (bf16)(acc[mi][ni][r] + bias);
      }
    }
}

// ---------------- GEMM 3: output projection -> f32 [M][H] ----------------
__global__ void __launch_bounds__(256) k_oproj(const bf16* __restrict__ CtxH,
                                               const bf16* __restrict__ WoT,
                                               const float* __restrict__ bo,
                                               float* __restrict__ CtxO) {
  const int lane = threadIdx.x & 31;
  const int w = blockIdx.x * 8 + (threadIdx.x >> 5);
  const int m0 = (w / 12) * 32;
  const int n0 = (w % 12) * 64;
  v8f acc[2][4];
#pragma unroll
  for (int i = 0; i < 2; ++i)
#pragma unroll
    for (int j = 0; j < 4; ++j) acc[i][j] = vzero();
  gemm32x64(CtxH, H_, WoT, H_, m0, n0, H_, lane, acc);
  const int hi = lane >> 4, rA = lane & 15;
#pragma unroll
  for (int mi = 0; mi < 2; ++mi)
#pragma unroll
    for (int ni = 0; ni < 4; ++ni) {
      const int n = n0 + ni * 16 + rA;
      const float bias = bo[n];
#pragma unroll
      for (int r = 0; r < 8; ++r) {
        const int m = m0 + mi * 16 + r + hi * 8;
        CtxO[(size_t)m * H_ + n] = acc[mi][ni][r] + bias;
      }
    }
}

// ---------------- Fused attention: scores -> softmax -> attn*V ----------------
__global__ void __launch_bounds__(256) k_attn(const bf16* __restrict__ Qh,
                                              const bf16* __restrict__ Kt,
                                              const bf16* __restrict__ Vt,
                                              float* __restrict__ attnP,
                                              bf16* __restrict__ CtxH) {
  __shared__ bf16 lds[8][16 * 72];                     // per-wave tile, stride 72 avoids bank conflicts
  const int lane = threadIdx.x & 31, wv = threadIdx.x >> 5;
  const int lin = blockIdx.x;                          // 16*8*32 = 4096
  const int b = lin >> 8, h = (lin >> 5) & 7, sg = lin & 31;
  const int s0 = (sg * 8 + wv) * 16;
  const int rA = lane & 15, hi = lane >> 4, k0 = hi * 8, kb = hi * 16;

  const bf16* qrow  = Qh + ((((size_t)b * NH_ + h) * S_) + s0 + rA) * HD_;
  const bf16* kbase = Kt + (size_t)h * C_ * HD_;

  v8f acc[4];
#pragma unroll
  for (int t = 0; t < 4; ++t) acc[t] = vzero();
#pragma unroll
  for (int nt = 0; nt < 4; ++nt) {
    const bf16* kcol = kbase + (size_t)(nt * 16 + rA) * HD_ + kb;
#pragma unroll
    for (int ks = 0; ks < HD_; ks += 32)
      acc[nt] = wmma_bf16(load_a(qrow + ks, k0), load_b(kcol + ks), acc[nt]);
  }

  const float scale = 0.10206207261596577f;            // 1/sqrt(96)
#pragma unroll
  for (int r = 0; r < 8; ++r) {
    float mx = -3.0e38f;
#pragma unroll
    for (int t = 0; t < 4; ++t) { acc[t][r] *= scale; mx = fmaxf(mx, acc[t][r]); }
#pragma unroll
    for (int d = 1; d < 16; d <<= 1) mx = fmaxf(mx, __shfl_xor(mx, d, 32));
    float sm = 0.f;
#pragma unroll
    for (int t = 0; t < 4; ++t) { float e = __expf(acc[t][r] - mx); acc[t][r] = e; sm += e; }
#pragma unroll
    for (int d = 1; d < 16; d <<= 1) sm += __shfl_xor(sm, d, 32);
    const float inv = 1.f / sm;
#pragma unroll
    for (int t = 0; t < 4; ++t) acc[t][r] *= inv;
  }

  const size_t arow = ((size_t)b * NH_ + h) * S_ + s0;
#pragma unroll
  for (int r = 0; r < 8; ++r) {
    const int row = r + hi * 8;
#pragma unroll
    for (int t = 0; t < 4; ++t) {
      const float p = acc[t][r];
      attnP[(arow + row) * C_ + t * 16 + rA] = p;      // f32 probs for attn_avg
      lds[wv][row * 72 + t * 16 + rA] = (bf16)p;       // re-layout C->A via LDS
    }
  }
  asm volatile("s_wait_dscnt 0" ::: "memory");         // in-wave LDS RAW fence

  const bf16* vbase = Vt + (size_t)h * HD_ * C_;
  v8f acc2[6];
#pragma unroll
  for (int t = 0; t < 6; ++t) acc2[t] = vzero();
#pragma unroll
  for (int nt = 0; nt < 6; ++nt) {
    const bf16* vcol = vbase + (size_t)(nt * 16 + rA) * C_ + kb;
#pragma unroll
    for (int ks = 0; ks < 64; ks += 32)
      acc2[nt] = wmma_bf16(load_a(&lds[wv][rA * 72 + ks], k0), load_b(vcol + ks), acc2[nt]);
  }
#pragma unroll
  for (int nt = 0; nt < 6; ++nt) {
    const int d = nt * 16 + rA;
#pragma unroll
    for (int r = 0; r < 8; ++r) {
      const int s = s0 + r + hi * 8;
      CtxH[((size_t)b * S_ + s) * H_ + h * HD_ + d] = (bf16)acc2[nt][r];
    }
  }
}

// ---------------- GEMM 4 (fused gate): relu(gate_in@Wg1+b) @ Wg2 -> sigmoid ----
// Block = 128 rows (8 waves x 16). B-panel (16 cols x 1536 K) staged in LDS by
// the Tensor Data Mover, double-buffered, shared by all 8 waves.
__global__ void __launch_bounds__(256) k_gate(const bf16* __restrict__ Xh,
                                              const float* __restrict__ CtxO,
                                              const bf16* __restrict__ Wg1T,
                                              const float* __restrict__ bg1,
                                              const float* __restrict__ relsum,
                                              const float* __restrict__ rel,
                                              const float* __restrict__ Wg2,
                                              const float* __restrict__ bg2,
                                              float* __restrict__ gate) {
  __shared__ __align__(16) char bpanel[2][TILE_PAD];   // 2 x 49920 B
  const int lane = threadIdx.x & 31, wv = threadIdx.x >> 5;
  const int m0 = blockIdx.x * 128 + wv * 16;
  const int rA = lane & 15, hi = lane >> 4, k0 = hi * 8, kb = hi * 16;
  const float rb = rel[m0 >> 12];                      // 128 | 4096 -> same batch
  const bf16*  xrow = Xh   + (size_t)(m0 + rA) * H_;
  const float* crow = CtxO + (size_t)(m0 + rA) * H_;
  const float bg2v = bg2[0];
  float rowacc[8] = {0.f,0.f,0.f,0.f,0.f,0.f,0.f,0.f};

#if USE_TDM
  if (threadIdx.x < 32)
    tdm_load_tile((unsigned int)(uintptr_t)&bpanel[0][0], Wg1T);
#endif
  for (int nt = 0; nt < 48; ++nt) {
    const int cur = nt & 1;
#if USE_TDM
    if (threadIdx.x < 32) {
      if (nt + 1 < 48) {
        tdm_load_tile((unsigned int)(uintptr_t)&bpanel[1 - cur][0],
                      Wg1T + (size_t)(nt + 1) * 16 * KG_);
        __builtin_amdgcn_s_wait_tensorcnt(1);          // tile nt landed (in-order)
      } else {
        __builtin_amdgcn_s_wait_tensorcnt(0);
      }
    }
    __syncthreads();
#else
    coop_load_tile(&bpanel[cur][0], Wg1T + (size_t)nt * 16 * KG_);
    __syncthreads();
#endif
    const char* bbase = &bpanel[cur][0];
    const int browb = rA * (KG_ * 2);                  // this lane's column base (bytes)
    v8f acc = vzero();
    for (int k = 0; k < H_; k += 32)                   // K = 0..767 : hidden (bf16)
      acc = wmma_bf16(load_a(xrow + k, k0),
                      load_b_lds(bbase, browb + (k + kb) * 2), acc);
    for (int k = 0; k < H_; k += 32)                   // K = 768..1535 : context (f32->bf16)
      acc = wmma_bf16(cvt_a_f32(crow + k, k0),
                      load_b_lds(bbase, browb + (H_ + k + kb) * 2), acc);
    const int n = nt * 16 + rA;
    const float bias = bg1[n] + rb * relsum[n];        // rank-1 rel_feat contribution
    const float w2 = Wg2[n];
#pragma unroll
    for (int r = 0; r < 8; ++r) {
      float t = acc[r] + bias;
      t = fmaxf(t, 0.f);                               // relu
      rowacc[r] += t * w2;                             // fold @ Wg2 immediately
    }
    __syncthreads();                                   // panel reuse fence
  }
#pragma unroll
  for (int r = 0; r < 8; ++r)
#pragma unroll
    for (int d = 1; d < 16; d <<= 1) rowacc[r] += __shfl_xor(rowacc[r], d, 32);
  if (rA == 0) {
#pragma unroll
    for (int r = 0; r < 8; ++r) {
      const int m = m0 + r + hi * 8;
      gate[m] = 1.f / (1.f + __expf(-(rowacc[r] + bg2v)));
    }
  }
}

// ---------------- residual + gate + LayerNorm (wave per row) ----------------
__global__ void __launch_bounds__(256) k_outln(const float* __restrict__ hid,
                                               const float* __restrict__ ctx,
                                               const float* __restrict__ gate,
                                               const float* __restrict__ ln_g,
                                               const float* __restrict__ ln_b,
                                               float* __restrict__ out) {
  const int lane = threadIdx.x & 31;
  const size_t row = (size_t)blockIdx.x * 8 + (threadIdx.x >> 5);
  const float g = gate[row];
  const float* hp = hid + row * H_;
  const float* cp = ctx + row * H_;
  float v[24], s = 0.f, s2 = 0.f;
#pragma unroll
  for (int i = 0; i < 24; ++i) {
    float x = hp[lane + i * 32] + g * cp[lane + i * 32];
    v[i] = x; s += x; s2 += x * x;
  }
#pragma unroll
  for (int d = 1; d < 32; d <<= 1) { s += __shfl_xor(s, d, 32); s2 += __shfl_xor(s2, d, 32); }
  const float mu = s * (1.f / 768.f);
  const float var = s2 * (1.f / 768.f) - mu * mu;
  const float rstd = rsqrtf(var + 1e-5f);
  float* op = out + row * H_;
#pragma unroll
  for (int i = 0; i < 24; ++i) {
    const int c = lane + i * 32;
    op[c] = (v[i] - mu) * rstd * ln_g[c] + ln_b[c];
  }
}

// ---------------- small prep / reduction kernels ----------------
__global__ void k_cast(const float* __restrict__ x, bf16* __restrict__ y, size_t n) {
  size_t i = (size_t)blockIdx.x * blockDim.x + threadIdx.x;
  const size_t st = (size_t)gridDim.x * blockDim.x;
  for (; i < n; i += st) y[i] = (bf16)x[i];
}

__global__ void k_tcast(const float* __restrict__ src, bf16* __restrict__ dst, int K, int N) {
  const size_t tid = (size_t)blockIdx.x * blockDim.x + threadIdx.x;
  if (tid >= (size_t)K * N) return;
  const int n = (int)(tid / K), k = (int)(tid % K);
  dst[tid] = (bf16)src[(size_t)k * N + n];             // dst[n][k] = src[k][n]
}

__global__ void k_relsum(const float* __restrict__ Wg1, float* __restrict__ relsum) {
  const int n = blockIdx.x * blockDim.x + threadIdx.x;
  if (n >= H_) return;
  float s = 0.f;
  for (int k = 2 * H_; k < 3 * H_; ++k) s += Wg1[(size_t)k * H_ + n];
  relsum[n] = s;
}

__global__ void k_kv(const float* __restrict__ conc,
                     const float* __restrict__ Wk, const float* __restrict__ bk,
                     const float* __restrict__ Wv, const float* __restrict__ bv,
                     bf16* __restrict__ Kt, bf16* __restrict__ Vt) {
  const int tid = blockIdx.x * blockDim.x + threadIdx.x;
  if (tid >= C_ * H_) return;
  const int c = tid / H_, n = tid % H_;
  float ka = bk[n], va = bv[n];
  for (int k = 0; k < H_; ++k) {
    const float e = conc[(size_t)c * H_ + k];
    ka += e * Wk[(size_t)k * H_ + n];
    va += e * Wv[(size_t)k * H_ + n];
  }
  const int h = n / HD_, d = n % HD_;
  Kt[((size_t)h * C_ + c) * HD_ + d] = (bf16)ka;       // [NH][C][HD] : B-frag for scores
  Vt[((size_t)h * HD_ + d) * C_ + c] = (bf16)va;       // [NH][HD][C] : B-frag for attn*V
}

__global__ void k_cpool(const float* __restrict__ conc, float* __restrict__ cp,
                        float* __restrict__ cpn) {
  __shared__ float red[768];
  const int n = threadIdx.x;
  float s = 0.f;
  for (int c = 0; c < C_; ++c) s += conc[(size_t)c * H_ + n];
  s *= (1.f / C_);
  cp[n] = s;
  red[n] = s * s;
  __syncthreads();
  for (int st = 384; st > 0; st >>= 1) {
    if (n < st) red[n] += red[n + st];
    __syncthreads();
  }
  if (n == 0) cpn[0] = sqrtf(red[0]);
}

__global__ void k_tpool(const float* __restrict__ hid, float* __restrict__ tp) {
  const int b = blockIdx.x / 3;
  const int col = (blockIdx.x % 3) * 256 + threadIdx.x;
  const float* p = hid + (size_t)b * S_ * H_ + col;
  float s = 0.f;
  for (int ss = 0; ss < S_; ++ss) s += p[(size_t)ss * H_];
  tp[(size_t)b * H_ + col] = s * (1.f / S_);
}

__global__ void k_rel(const float* __restrict__ tp, const float* __restrict__ cp,
                      const float* __restrict__ cpn, float* __restrict__ rel) {
  __shared__ float rd[256], rn[256];
  for (int b = 0; b < B_; ++b) {
    float d = 0.f, n2 = 0.f;
    for (int i = threadIdx.x; i < H_; i += 256) {
      const float t = tp[(size_t)b * H_ + i];
      d += t * cp[i]; n2 += t * t;
    }
    rd[threadIdx.x] = d; rn[threadIdx.x] = n2;
    __syncthreads();
    for (int st = 128; st > 0; st >>= 1) {
      if (threadIdx.x < st) { rd[threadIdx.x] += rd[threadIdx.x + st]; rn[threadIdx.x] += rn[threadIdx.x + st]; }
      __syncthreads();
    }
    if (threadIdx.x == 0)
      rel[b] = rd[0] / (fmaxf(sqrtf(rn[0]), 1e-8f) * fmaxf(cpn[0], 1e-8f));
    __syncthreads();
  }
}

__global__ void k_attnavg(const float* __restrict__ attnP, float* __restrict__ out2) {
  const size_t tid = (size_t)blockIdx.x * blockDim.x + threadIdx.x;
  if (tid >= (size_t)M_ * C_) return;
  const size_t m = tid >> 6;
  const int c = (int)(tid & 63);
  const size_t b = m >> 12, s = m & 4095;
  const float* p = attnP + ((b * NH_) * S_ + s) * C_ + c;
  float sum = 0.f;
#pragma unroll
  for (int h = 0; h < NH_; ++h) sum += p[(size_t)h * S_ * C_];
  out2[tid] = sum * 0.125f;
}

// ---------------- host launch ----------------
extern "C" void kernel_launch(void* const* d_in, const int* in_sizes, int n_in,
                              void* d_out, int out_size, void* d_ws, size_t ws_size,
                              hipStream_t stream) {
  const float* hid  = (const float*)d_in[0];
  const float* conc = (const float*)d_in[1];
  const float* Wq   = (const float*)d_in[2];
  const float* bq   = (const float*)d_in[3];
  const float* Wk   = (const float*)d_in[4];
  const float* bk   = (const float*)d_in[5];
  const float* Wv   = (const float*)d_in[6];
  const float* bv   = (const float*)d_in[7];
  const float* Wo   = (const float*)d_in[8];
  const float* bo   = (const float*)d_in[9];
  const float* Wg1  = (const float*)d_in[10];
  const float* bg1  = (const float*)d_in[11];
  const float* Wg2  = (const float*)d_in[12];
  const float* bg2  = (const float*)d_in[13];
  const float* ln_g = (const float*)d_in[14];
  const float* ln_b = (const float*)d_in[15];

  char* ws = (char*)d_ws;
  size_t off = 0;
  auto alloc = [&](size_t bytes) -> char* {
    char* p = ws + off;
    off += (bytes + 255) & ~(size_t)255;
    return p;
  };
  bf16*  Xh    = (bf16*) alloc((size_t)M_ * H_ * 2);          // hidden bf16
  bf16*  Qh    = (bf16*) alloc((size_t)M_ * H_ * 2);          // Q [B][NH][S][HD]
  bf16*  CtxH  = (bf16*) alloc((size_t)M_ * H_ * 2);          // attn*V (pre-Wo) bf16
  float* CtxO  = (float*)alloc((size_t)M_ * H_ * 4);          // context after Wo
  float* attnP = (float*)alloc((size_t)M_ * NH_ * C_ * 4);    // per-head probs
  bf16*  WqT   = (bf16*) alloc((size_t)H_ * H_ * 2);
  bf16*  WoT   = (bf16*) alloc((size_t)H_ * H_ * 2);
  bf16*  Wg1T  = (bf16*) alloc((size_t)H_ * KG_ * 2);
  bf16*  Kt    = (bf16*) alloc((size_t)NH_ * C_ * HD_ * 2);
  bf16*  Vt    = (bf16*) alloc((size_t)NH_ * HD_ * C_ * 2);
  float* cp    = (float*)alloc(H_ * 4);
  float* cpn   = (float*)alloc(256);
  float* tp    = (float*)alloc((size_t)B_ * H_ * 4);
  float* rel   = (float*)alloc(256);
  float* rsum  = (float*)alloc(H_ * 4);
  float* gate  = (float*)alloc((size_t)M_ * 4);

  float* out  = (float*)d_out;
  float* out2 = out + (size_t)M_ * H_;

  // prep
  k_cast  <<<8192, 256, 0, stream>>>(hid, Xh, (size_t)M_ * H_);
  k_tcast <<<(H_ * H_ + 255) / 256, 256, 0, stream>>>(Wq, WqT, H_, H_);
  k_tcast <<<(H_ * H_ + 255) / 256, 256, 0, stream>>>(Wo, WoT, H_, H_);
  k_tcast <<<(H_ * KG_ + 255) / 256, 256, 0, stream>>>(Wg1, Wg1T, KG_, H_);
  k_relsum<<<3, 256, 0, stream>>>(Wg1, rsum);
  k_kv    <<<(C_ * H_ + 255) / 256, 256, 0, stream>>>(conc, Wk, bk, Wv, bv, Kt, Vt);
  k_cpool <<<1, 768, 0, stream>>>(conc, cp, cpn);
  k_tpool <<<B_ * 3, 256, 0, stream>>>(hid, tp);
  k_rel   <<<1, 256, 0, stream>>>(tp, cp, cpn, rel);

  // WMMA pipeline
  k_qproj <<<3072, 256, 0, stream>>>(Xh, WqT, bq, Qh);
  k_attn  <<<4096, 256, 0, stream>>>(Qh, Kt, Vt, attnP, CtxH);
  k_oproj <<<3072, 256, 0, stream>>>(CtxH, WoT, bo, CtxO);
  k_gate  <<<512, 256, 0, stream>>>(Xh, CtxO, Wg1T, bg1, rsum, rel, Wg2, bg2, gate);

  // epilogues
  k_outln <<<8192, 256, 0, stream>>>(hid, CtxO, gate, ln_g, ln_b, out);
  k_attnavg<<<((size_t)M_ * C_ + 255) / 256, 256, 0, stream>>>(attnP, out2);

  (void)in_sizes; (void)n_in; (void)out_size; (void)ws_size;
}
